// AxialAttention_44375602102349
// MI455X (gfx1250) — compile-verified, matrix operand order checked by reading, not run on hardware
//
#include <hip/hip_runtime.h>

typedef __attribute__((ext_vector_type(16))) _Float16 v16h;
typedef __attribute__((ext_vector_type(8)))  _Float16 v8h;
typedef __attribute__((ext_vector_type(2)))  _Float16 v2h;
typedef __attribute__((ext_vector_type(8)))  float    v8f;
typedef int v4i __attribute__((vector_size(16)));

#define CDIM   512
#define HEADS  8
#define DHEAD  64
#define HH     96
#define WW     96
#define PIX    (HH * WW)          // 9216 pixels per batch
#define BATCH  4
#define SCALE  0.125f

// ---- async global->LDS (gfx1250 GLOBAL_LOAD_ASYNC_TO_LDS_B128, ASYNCcnt) ----
#if __has_builtin(__builtin_amdgcn_global_load_async_to_lds_b128)
#define HAVE_ASYNC_LDS 1
__device__ __forceinline__ void async_copy_b128(void* lds_dst, const void* gsrc) {
    __builtin_amdgcn_global_load_async_to_lds_b128(
        (__attribute__((address_space(1))) v4i*)gsrc,
        (__attribute__((address_space(3))) v4i*)lds_dst,
        0, 0);
}
#else
#define HAVE_ASYNC_LDS 0
#endif

__device__ __forceinline__ void wait_async0() {
#if __has_builtin(__builtin_amdgcn_s_wait_asynccnt)
    __builtin_amdgcn_s_wait_asynccnt(0);
#else
    asm volatile("s_wait_asynccnt 0" ::: "memory");
#endif
}

union V16U { v16h v; v8h h[2]; };

// Load a 16-element A/B fragment whose K-runs are contiguous in memory.
// p points at (row base + k_block*32). Lane half selects K offset 0 or 8;
// elements 0..7 = K run at +0, elements 8..15 = K run at +16 (ISA 7.12.2).
__device__ __forceinline__ v16h load_frag(const _Float16* p, int lane) {
    const _Float16* q = p + ((lane < 16) ? 0 : 8);
    V16U u;
    u.h[0] = *(const v8h*)q;
    u.h[1] = *(const v8h*)(q + 16);
    return u.v;
}

__device__ __forceinline__ v8f wmma16(v16h a, v16h b, v8f c) {
    return __builtin_amdgcn_wmma_f32_16x16x32_f16(false, a, false, b,
                                                  (short)0, c, false, false);
}

// ---------------------------------------------------------------------------
// Kernel 0: one-time f32 -> f16 conversion of both weight matrices
// ---------------------------------------------------------------------------
__global__ __launch_bounds__(256)
void cvt_weights_kernel(const float* __restrict__ wq,
                        const float* __restrict__ wo,
                        _Float16* __restrict__ wq16,
                        _Float16* __restrict__ wo16) {
    const int i = (blockIdx.x * 256 + threadIdx.x) * 4;
    if (i < 3 * CDIM * CDIM) {
#pragma unroll
        for (int j = 0; j < 4; ++j) wq16[i + j] = (_Float16)wq[i + j];
    } else {
        const int k = i - 3 * CDIM * CDIM;
#pragma unroll
        for (int j = 0; j < 4; ++j) wo16[k + j] = (_Float16)wo[k + j];
    }
}

// ---------------------------------------------------------------------------
// Kernel 1: QKV projection  qkv[b,o,p] = sum_c w_qkv[o,c] * x[b,c,p]
// grid (288 ntile, 12 mgroup, 4 batch), block 256 (8 waves)
// ---------------------------------------------------------------------------
__global__ __launch_bounds__(256)
void qkv_proj_kernel(const float* __restrict__ x,
                     const _Float16* __restrict__ wq16,
                     _Float16* __restrict__ qkv_ws) {
    __shared__ __align__(16) _Float16 xsT[32][CDIM];   // [pixel][k], 32 KB

    const int t     = threadIdx.x;
    const int ntile = blockIdx.x;
    const int b     = blockIdx.z;

    // cooperative load + f32->f16 convert, transposed into [n][k]
    {
        const int col = t & 31;
        const int pb  = ntile * 32 + col;
        const float* xb = x + (size_t)b * CDIM * PIX + pb;
        for (int c = (t >> 5) * 2; c < CDIM; c += 16) {
            v2h pair = { (_Float16)xb[(size_t)c * PIX],
                         (_Float16)xb[(size_t)(c + 1) * PIX] };
            *(v2h*)&xsT[col][c] = pair;
        }
    }
    __syncthreads();

    const int wave = t >> 5;
    const int lane = t & 31;
    const int m0   = blockIdx.y * 128 + wave * 16;
    const int nloc = lane & 15;
    const int moff = (lane < 16) ? 0 : 8;

    const _Float16* arow  = wq16 + (size_t)(m0 + nloc) * CDIM;
    const _Float16* brow0 = &xsT[nloc][0];
    const _Float16* brow1 = &xsT[nloc + 16][0];

    v8f acc0 = {}, acc1 = {};
    for (int kk = 0; kk < 16; ++kk) {
        const int kb = kk * 32;
        __builtin_prefetch(arow + kb + 128, 0, 1);
        const v16h af  = load_frag(arow + kb, lane);
        const v16h bf0 = load_frag(brow0 + kb, lane);
        const v16h bf1 = load_frag(brow1 + kb, lane);
        acc0 = wmma16(af, bf0, acc0);
        acc1 = wmma16(af, bf1, acc1);
    }

    // scatter to [b, s, head, pixel, d] f16 (d consecutive across v -> b128)
#pragma unroll
    for (int half = 0; half < 2; ++half) {
        const v8f acc = half ? acc1 : acc0;
        const int pbp = ntile * 32 + half * 16 + nloc;
#pragma unroll
        for (int v = 0; v < 8; ++v) {
            const int o    = m0 + v + moff;
            const int s    = o >> 9;
            const int rem  = o & 511;
            const int head = rem >> 6;
            const int d    = rem & 63;
            const size_t idx =
                ((((size_t)b * 3 + s) * HEADS + head) * PIX + pbp) * DHEAD + d;
            qkv_ws[idx] = (_Float16)acc[v];
        }
    }
}

// ---------------------------------------------------------------------------
// Kernel 2: axial attention over width, one block per (b, head, h)
// ---------------------------------------------------------------------------
__global__ __launch_bounds__(192)
void attn_kernel(const _Float16* __restrict__ qkv_ws,
                 _Float16* __restrict__ attn_ws) {
    __shared__ __align__(16) _Float16 kls[WW][DHEAD];   // [j][d], 12 KB
    __shared__ __align__(16) _Float16 vT [DHEAD][WW];   // [d][j], 12 KB
    __shared__ __align__(16) _Float16 pls[6][16][WW];   // per-wave P, 18 KB

    const int t    = threadIdx.x;
    const int h    = blockIdx.x;
    const int head = blockIdx.y;
    const int b    = blockIdx.z;

    const size_t strideS = (size_t)HEADS * PIX * DHEAD;
    const size_t base    = (((size_t)b * 3 * HEADS + head) * PIX + (size_t)h * WW) * DHEAD;
    const _Float16* qg = qkv_ws + base;
    const _Float16* kg = qkv_ws + base + strideS;
    const _Float16* vg = qkv_ws + base + 2 * strideS;

    // K: async DMA straight copy [j][d]; V: load + transpose into [d][j]
    for (int i8 = t * 8; i8 < WW * DHEAD; i8 += 192 * 8) {
        const int w = i8 >> 6, d = i8 & 63;
#if HAVE_ASYNC_LDS
        async_copy_b128(&kls[w][d], kg + i8);
#else
        *(v8h*)&kls[w][d] = *(const v8h*)(kg + i8);
#endif
        const v8h vv = *(const v8h*)(vg + i8);
#pragma unroll
        for (int j = 0; j < 8; ++j) vT[d + j][w] = vv[j];
    }
#if HAVE_ASYNC_LDS
    wait_async0();
#endif
    __syncthreads();

    const int wave = t >> 5;
    const int lane = t & 31;
    const int nloc = lane & 15;
    const int moff = (lane < 16) ? 0 : 8;
    const int i0   = wave * 16;

    // Q strip A-fragments (d contiguous -> vector loads)
    v16h qf[2];
    {
        const _Float16* qrow = qg + (size_t)(i0 + nloc) * DHEAD;
        qf[0] = load_frag(qrow, lane);
        qf[1] = load_frag(qrow + 32, lane);
    }

    // S = Q K^T
    v8f sacc[6];
#pragma unroll
    for (int jt = 0; jt < 6; ++jt) {
        v8f a = {};
        const _Float16* krow = &kls[jt * 16 + nloc][0];
        a = wmma16(qf[0], load_frag(krow, lane), a);
        a = wmma16(qf[1], load_frag(krow + 32, lane), a);
        sacc[jt] = a;
    }

    // row softmax (row r at vgpr r, wave halves hold r and r+8)
#pragma unroll
    for (int r = 0; r < 8; ++r) {
        float m = -3.0e38f;
#pragma unroll
        for (int jt = 0; jt < 6; ++jt) m = fmaxf(m, sacc[jt][r]);
        m = fmaxf(m, __shfl_xor(m, 1, 32));
        m = fmaxf(m, __shfl_xor(m, 2, 32));
        m = fmaxf(m, __shfl_xor(m, 4, 32));
        m = fmaxf(m, __shfl_xor(m, 8, 32));
        float s = 0.0f;
#pragma unroll
        for (int jt = 0; jt < 6; ++jt) {
            float p = __expf((sacc[jt][r] - m) * SCALE);
            sacc[jt][r] = p;
            s += p;
        }
        s += __shfl_xor(s, 1, 32);
        s += __shfl_xor(s, 2, 32);
        s += __shfl_xor(s, 4, 32);
        s += __shfl_xor(s, 8, 32);
        const float inv = 1.0f / s;
#pragma unroll
        for (int jt = 0; jt < 6; ++jt) sacc[jt][r] *= inv;
    }

    // C-layout -> A-layout via LDS
#pragma unroll
    for (int jt = 0; jt < 6; ++jt)
#pragma unroll
        for (int v = 0; v < 8; ++v)
            pls[wave][v + moff][jt * 16 + nloc] = (_Float16)sacc[jt][v];
    __syncthreads();

    // O = P V   (A from pls rows, B from vT rows -- both k-contiguous)
    v8f oacc[4];
    const _Float16* prow = &pls[wave][nloc][0];
#pragma unroll
    for (int dt = 0; dt < 4; ++dt) {
        v8f a = {};
        const _Float16* vrow = &vT[dt * 16 + nloc][0];
#pragma unroll
        for (int js = 0; js < 3; ++js)
            a = wmma16(load_frag(prow + js * 32, lane),
                       load_frag(vrow + js * 32, lane), a);
        oacc[dt] = a;
    }

    // store to attn ws [b*PIX + pixel][c = head*64 + d] f16
#pragma unroll
    for (int dt = 0; dt < 4; ++dt)
#pragma unroll
        for (int v = 0; v < 8; ++v) {
            const int wpos = i0 + v + moff;
            const int d    = dt * 16 + nloc;
            const size_t idx = ((size_t)b * PIX + (size_t)h * WW + wpos) * CDIM
                             + head * DHEAD + d;
            attn_ws[idx] = (_Float16)oacc[dt][v];
        }
}

// ---------------------------------------------------------------------------
// Kernel 3: out projection  out[b,o,p] = sum_c w_out[o,c]*attn[b,p,c] + b_out[o]
// grid (288 ntile, 4 mgroup, 4 batch), block 256
// ---------------------------------------------------------------------------
__global__ __launch_bounds__(256)
void out_proj_kernel(const _Float16* __restrict__ attn_ws,
                     const _Float16* __restrict__ wo16,
                     const float* __restrict__ bout,
                     float* __restrict__ out) {
    __shared__ __align__(16) _Float16 asT[32][CDIM];   // [pixel][c], 32 KB

    const int t     = threadIdx.x;
    const int ntile = blockIdx.x;
    const int b     = blockIdx.z;

    {
        const int col = t & 31;
        const int pb  = ntile * 32 + col;
        const _Float16* arow = attn_ws + ((size_t)b * PIX + pb) * CDIM;
        for (int c = (t >> 5) * 8; c < CDIM; c += 64) {
#if HAVE_ASYNC_LDS
            async_copy_b128(&asT[col][c], arow + c);
#else
            *(v8h*)&asT[col][c] = *(const v8h*)(arow + c);
#endif
        }
    }
#if HAVE_ASYNC_LDS
    wait_async0();
#endif
    __syncthreads();

    const int wave = t >> 5;
    const int lane = t & 31;
    const int m0   = blockIdx.y * 128 + wave * 16;
    const int nloc = lane & 15;
    const int moff = (lane < 16) ? 0 : 8;

    const _Float16* arow  = wo16 + (size_t)(m0 + nloc) * CDIM;
    const _Float16* brow0 = &asT[nloc][0];
    const _Float16* brow1 = &asT[nloc + 16][0];

    v8f acc0 = {}, acc1 = {};
    for (int kk = 0; kk < 16; ++kk) {
        const int kb = kk * 32;
        __builtin_prefetch(arow + kb + 128, 0, 1);
        const v16h af = load_frag(arow + kb, lane);
        acc0 = wmma16(af, load_frag(brow0 + kb, lane), acc0);
        acc1 = wmma16(af, load_frag(brow1 + kb, lane), acc1);
    }

#pragma unroll
    for (int half = 0; half < 2; ++half) {
        const v8f acc = half ? acc1 : acc0;
        const int pb  = ntile * 32 + half * 16 + nloc;
#pragma unroll
        for (int v = 0; v < 8; ++v) {
            const int o = m0 + v + moff;
            out[((size_t)b * CDIM + o) * PIX + pb] = acc[v] + bout[o];
        }
    }
}

// ---------------------------------------------------------------------------
extern "C" void kernel_launch(void* const* d_in, const int* in_sizes, int n_in,
                              void* d_out, int out_size, void* d_ws, size_t ws_size,
                              hipStream_t stream) {
    const float* x    = (const float*)d_in[0];
    const float* wqkv = (const float*)d_in[1];
    const float* wout = (const float*)d_in[2];
    const float* bout = (const float*)d_in[3];
    float* out = (float*)d_out;

    _Float16* qkv_ws  = (_Float16*)d_ws;                               // 56,623,104
    _Float16* attn_ws = qkv_ws + (size_t)BATCH * 3 * CDIM * PIX;       // 18,874,368
    _Float16* wq16    = attn_ws + (size_t)BATCH * PIX * CDIM;          //    786,432
    _Float16* wo16    = wq16 + (size_t)3 * CDIM * CDIM;                //    262,144

    cvt_weights_kernel<<<dim3((3 * CDIM * CDIM + CDIM * CDIM) / 1024), 256, 0, stream>>>(
        wqkv, wout, wq16, wo16);
    qkv_proj_kernel<<<dim3(PIX / 32, 12, BATCH), 256, 0, stream>>>(x, wq16, qkv_ws);
    attn_kernel    <<<dim3(HH, HEADS, BATCH),    192, 0, stream>>>(qkv_ws, attn_ws);
    out_proj_kernel<<<dim3(PIX / 32, 4, BATCH),  256, 0, stream>>>(attn_ws, wo16, bout, out);
}
